// MS_SSA_Conv_9552007266868
// MI455X (gfx1250) — matrix operand order repaired and checked by reading, not compiled, not run on hardware
//
#include <hip/hip_runtime.h>

typedef __attribute__((ext_vector_type(16))) _Float16 v16h;
typedef __attribute__((ext_vector_type(8)))  _Float16 v8h;
typedef __attribute__((ext_vector_type(8)))  float    v8f;
typedef __attribute__((ext_vector_type(4)))  unsigned int ui4;
typedef __attribute__((ext_vector_type(8)))  int i8v;
typedef __attribute__((ext_vector_type(4)))  int i4v;

#define T_ 4
#define B_ 8
#define C_ 384
#define N_ 1024
#define HEADS_ 8
#define HD_ 48

// TDM-padded LDS panel geometry: row = 384 halves = 768B data, +16B pad per 256B
// -> row stride 816B; byte(h, R) = 816*R + 2*h + 16*(h>>7)
#define ROWB_ 816

// workspace layout (bytes)
#define XS_OFF    ((size_t)0)           // [T,B,N,C] f16 spikes (reused as masked-q later)
#define Q_OFF     ((size_t)25165824)    // [T,B,N,C] f16
#define K_OFF     ((size_t)50331648)
#define V_OFF     ((size_t)75497472)
#define W_OFF     ((size_t)100663296)   // 4 x [384,384] f16 BN-folded weights (q,k,v,proj)
#define BIAS_OFF  ((size_t)101842944)   // 4 x 384 f32 folded biases
#define KVRAW_OFF ((size_t)101849088)   // [T*B,384] f32
#define KVS_OFF   ((size_t)101898240)   // [T*B,384] f32 kv spikes

// ---------------------------------------------------------------------------
// Kernel 0: fold BN (and proj bias) into f16 weights + f32 biases.
// ---------------------------------------------------------------------------
__global__ __launch_bounds__(256) void prep_weights(
    const float* wq, const float* wk, const float* wv, const float* wp,
    const float* projb,
    const float* gq, const float* beq, const float* mq, const float* vq,
    const float* gk, const float* bek, const float* mk, const float* vk,
    const float* gv, const float* bev, const float* mv, const float* vv,
    const float* gp, const float* bep, const float* mp, const float* vp,
    _Float16* __restrict__ wout, float* __restrict__ biasout)
{
  int gid = blockIdx.x * 256 + threadIdx.x;
  if (gid >= 4 * C_ * C_) return;
  int m = gid / (C_ * C_);
  int r = gid % (C_ * C_);
  int o = r / C_;
  int c = r % C_;
  const float *W, *g, *be, *mean, *var;
  switch (m) {
    case 0:  W = wq; g = gq; be = beq; mean = mq; var = vq; break;
    case 1:  W = wk; g = gk; be = bek; mean = mk; var = vk; break;
    case 2:  W = wv; g = gv; be = bev; mean = mv; var = vv; break;
    default: W = wp; g = gp; be = bep; mean = mp; var = vp; break;
  }
  float scale = g[o] * rsqrtf(var[o] + 1e-5f);
  wout[(size_t)m * C_ * C_ + r] = (_Float16)(W[r] * scale);
  if (c == 0) {
    float bb = (m == 3) ? (be[o] + (projb[o] - mean[o]) * scale)
                        : (be[o] - mean[o] * scale);
    biasout[m * C_ + o] = bb;
  }
}

// ---------------------------------------------------------------------------
// Kernel 1: shortcut LIF over T + transpose [T,B,C,N] -> [T,B,N,C] f16 spikes.
// ---------------------------------------------------------------------------
__global__ __launch_bounds__(256) void lif_transpose(const float* __restrict__ x,
                                                     _Float16* __restrict__ xs)
{
  __shared__ _Float16 tile[T_][32][33];
  int n0 = blockIdx.x * 32;
  int c0 = blockIdx.y * 32;
  int b  = blockIdx.z;
  int tx = threadIdx.x;
  int ty = threadIdx.y;
  for (int i = 0; i < 4; ++i) {
    int cl = ty + i * 8;
    int c  = c0 + cl;
    float vm = 0.f;
    #pragma unroll
    for (int t = 0; t < T_; ++t) {
      float xv = x[(((size_t)t * B_ + b) * C_ + c) * N_ + n0 + tx];
      vm = 0.5f * (vm + xv);
      float s = (vm >= 0.5f) ? 1.f : 0.f;
      vm = (s > 0.f) ? 0.f : vm;
      tile[t][cl][tx] = (_Float16)s;
    }
  }
  __syncthreads();
  int tid = ty * 32 + tx;
  int cc = tid & 31;
  int nn = tid >> 5;
  for (int i = 0; i < 4; ++i) {
    int nl = nn + i * 8;
    #pragma unroll
    for (int t = 0; t < T_; ++t) {
      xs[(((size_t)t * B_ + b) * N_ + n0 + nl) * C_ + c0 + cc] = tile[t][cc][nl];
    }
  }
}

// ---------------------------------------------------------------------------
// Kernel 2: fused Q/K/V GEMM + folded BN bias + LIF.
// Workgroup = (b, 16-pixel n-tile). Wave 0 issues ONE tensor_load_to_lds that
// DMAs xs[t=0..3][b][n0..n0+15][0..383] into LDS (3-D tile: x=384 ch,
// y=16 rows stride 384, z=4 timesteps stride B*N*C; pad 4 DW / 64 DW for
// bank-conflict-free rows). 8 waves then run 9 rounds covering the 72
// (matrix, ctile) tasks, B-fragments from LDS, LIF state in accumulators.
// ---------------------------------------------------------------------------
__global__ __launch_bounds__(256) void qkv_gemm(const _Float16* __restrict__ xs,
                                                const _Float16* __restrict__ wfold,
                                                const float* __restrict__ biasfold,
                                                _Float16* __restrict__ qbuf,
                                                _Float16* __restrict__ kbuf,
                                                _Float16* __restrict__ vbuf)
{
  __shared__ __align__(16) unsigned char ldsbuf[4 * 16 * ROWB_];   // 52224 B
  int nt = blockIdx.x & 63;
  int b  = blockIdx.x >> 6;
  int wv   = threadIdx.x >> 5;
  int lane = threadIdx.x & 31;
  int lr = lane & 15, khi = lane >> 4;

  if (wv == 0) {
    unsigned long long ga =
        (unsigned long long)(uintptr_t)(xs + ((size_t)b * N_ + (size_t)nt * 16) * C_);
    unsigned int lb32 = (unsigned int)(uintptr_t)ldsbuf;  // low 32 bits = LDS offset
    ui4 g0; i8v g1; i4v g2; i4v g3 = {};
    g0[0] = 1u;                                   // count=1, user descriptor
    g0[1] = lb32;                                 // lds_addr
    g0[2] = (unsigned int)ga;                     // global_addr[31:0]
    g0[3] = (unsigned int)(ga >> 32) | (2u << 30);// global_addr[56:32] | type=2
    g1[0] = (1 << 16) | (1 << 20) | (5 << 22) | (3 << 25);
           // data_size=2B, pad_enable, pad_interval=64DW, pad_amount=4DW
    g1[1] = (384u & 0xFFFFu) << 16;               // tensor_dim0 lo16 (=384)
    g1[2] = (unsigned int)N_ << 16;               // dim0 hi16=0 | tensor_dim1 lo16=1024
    g1[3] = (384 << 16);                          // dim1 hi16=0 | tile_dim0=384
    g1[4] = 16 | (4 << 16);                       // tile_dim1=16 | tile_dim2=4
    g1[5] = 384;                                  // tensor_dim0_stride lo32
    g1[6] = 0;                                    // stride0 hi16 | stride1 lo16 (0x300000 lo16=0)
    g1[7] = (int)((unsigned)(B_ * N_ * C_) >> 16);// tensor_dim1_stride[47:16] = 0x30
    g2[0] = 4; g2[1] = 0; g2[2] = 0; g2[3] = 0;   // tensor_dim2=4
    asm volatile("tensor_load_to_lds %0, %1, %2, %3"
                 :: "s"(g0), "s"(g1), "s"(g2), "s"(g3)
                 : "memory");
    __builtin_amdgcn_s_wait_tensorcnt(0);
  }
  __syncthreads();

  const unsigned char* lb = ldsbuf;

  for (int round = 0; round < 9; ++round) {
    int task = wv * 9 + round;                    // 0..71
    int m = task / 24, ct = task % 24;
    const _Float16* wrow = wfold + (size_t)m * C_ * C_ + (size_t)(ct * 16 + lr) * C_;
    const float*    Bm   = biasfold + m * C_;
    _Float16* dst = (m == 0) ? qbuf : (m == 1) ? kbuf : vbuf;

    if (round < 8) {                              // prefetch next round's A rows
      int t2 = wv * 9 + round + 1;
      __builtin_prefetch(wfold + (size_t)(t2 / 24) * C_ * C_ +
                         (size_t)((t2 % 24) * 16 + lr) * C_, 0, 1);
    }

    v16h afr[12];
    #pragma unroll
    for (int kk = 0; kk < 12; ++kk) {
      int kb = kk * 32;
      float4* ap = reinterpret_cast<float4*>(&afr[kk]);
      ap[0] = *reinterpret_cast<const float4*>(wrow + kb + 16 * khi);
      ap[1] = *reinterpret_cast<const float4*>(wrow + kb + 16 * khi + 8);
    }
    v8f bias;
    #pragma unroll
    for (int r = 0; r < 8; ++r) bias[r] = Bm[ct * 16 + r + 8 * khi];

    v8f vm = {};
    #pragma unroll
    for (int t = 0; t < T_; ++t) {
      int rb = ROWB_ * (t * 16 + lr);
      v8f acc = {};
      #pragma unroll
      for (int kk = 0; kk < 12; ++kk) {
        int h   = kk * 32 + 16 * khi;
        int off = rb + 2 * h + 16 * (h >> 7);     // TDM-padded layout
        v16h bfr;
        float4* bp = reinterpret_cast<float4*>(&bfr);
        bp[0] = *reinterpret_cast<const float4*>(lb + off);
        bp[1] = *reinterpret_cast<const float4*>(lb + off + 16);
        acc = __builtin_amdgcn_wmma_f32_16x16x32_f16(false, afr[kk], false, bfr,
                                                     (short)0, acc, false, false);
      }
      v8h sp;
      #pragma unroll
      for (int r = 0; r < 8; ++r) {
        float y  = acc[r] + bias[r];
        float nv = 0.5f * (vm[r] + y);
        float s  = (nv >= 0.5f) ? 1.f : 0.f;
        vm[r] = (s > 0.f) ? 0.f : nv;
        sp[r] = (_Float16)s;
      }
      _Float16* drow = dst + (((size_t)t * B_ + b) * N_ + nt * 16 + lr) * C_
                           + ct * 16 + 8 * khi;
      *reinterpret_cast<v8h*>(drow) = sp;
    }
  }
}

// ---------------------------------------------------------------------------
// Kernel 3: kv[t,b,c] = sum_n k[t,b,n,c] * v[t,b,n,c]
// ---------------------------------------------------------------------------
__global__ __launch_bounds__(384) void kv_reduce(const _Float16* __restrict__ kbuf,
                                                 const _Float16* __restrict__ vbuf,
                                                 float* __restrict__ kvraw)
{
  int tb = blockIdx.x;
  int c  = threadIdx.x;
  const _Float16* kp = kbuf + (size_t)tb * N_ * C_ + c;
  const _Float16* vp = vbuf + (size_t)tb * N_ * C_ + c;
  float acc = 0.f;
  for (int n = 0; n < N_; ++n)
    acc += (float)kp[(size_t)n * C_] * (float)vp[(size_t)n * C_];
  kvraw[tb * C_ + c] = acc;
}

// ---------------------------------------------------------------------------
// Kernel 4: talking heads + LIF over T -> kv spikes.
// ---------------------------------------------------------------------------
__global__ __launch_bounds__(384) void talk_lif(const float* __restrict__ thw,
                                                const float* __restrict__ kvraw,
                                                float* __restrict__ kvs)
{
  int b = blockIdx.x;
  int c = threadIdx.x;
  int o = c / HD_, d = c % HD_;
  float vm = 0.f;
  #pragma unroll
  for (int t = 0; t < T_; ++t) {
    float y = 0.f;
    #pragma unroll
    for (int h = 0; h < HEADS_; ++h)
      y += thw[o * HEADS_ + h] * kvraw[(t * B_ + b) * C_ + h * HD_ + d];
    vm = 0.5f * (vm + y);
    float s = (vm >= 0.5f) ? 1.f : 0.f;
    vm = (s > 0.f) ? 0.f : vm;
    kvs[(t * B_ + b) * C_ + c] = s;
  }
}

// ---------------------------------------------------------------------------
// Kernel 5: x_att = q * kv_spike (per-channel gate).
// ---------------------------------------------------------------------------
__global__ __launch_bounds__(256) void mask_q(const _Float16* __restrict__ qbuf,
                                              const float* __restrict__ kvs,
                                              _Float16* __restrict__ qm)
{
  size_t gid = (size_t)blockIdx.x * 256 + threadIdx.x;
  size_t total = (size_t)T_ * B_ * N_ * C_;
  if (gid >= total) return;
  int c  = (int)(gid % C_);
  int tb = (int)(gid / ((size_t)N_ * C_));
  float msk = kvs[tb * C_ + c];
  qm[gid] = (msk > 0.5f) ? qbuf[gid] : (_Float16)0.0f;
}

// ---------------------------------------------------------------------------
// Kernel 6: proj GEMM + folded bias/BN + identity residual.
// Workgroup = (tb, n-tile); one 2-D tensor_load_to_lds stages the 16x384
// masked-q panel; 8 waves x 3 rounds cover the 24 ctiles from LDS.
// ---------------------------------------------------------------------------
__global__ __launch_bounds__(256) void proj_gemm(const _Float16* __restrict__ qm,
                                                 const _Float16* __restrict__ wfold,
                                                 const float* __restrict__ biasfold,
                                                 const float* __restrict__ x,
                                                 float* __restrict__ out)
{
  __shared__ __align__(16) unsigned char lds2[16 * ROWB_];   // 13056 B
  int nt = blockIdx.x & 63;
  int tb = blockIdx.x >> 6;                    // 0..31
  int wv   = threadIdx.x >> 5;
  int lane = threadIdx.x & 31;
  int lr = lane & 15, khi = lane >> 4;

  if (wv == 0) {
    unsigned long long ga =
        (unsigned long long)(uintptr_t)(qm + ((size_t)tb * N_ + (size_t)nt * 16) * C_);
    unsigned int lb32 = (unsigned int)(uintptr_t)lds2;
    ui4 g0; i8v g1;
    g0[0] = 1u;
    g0[1] = lb32;
    g0[2] = (unsigned int)ga;
    g0[3] = (unsigned int)(ga >> 32) | (2u << 30);
    g1[0] = (1 << 16) | (1 << 20) | (5 << 22) | (3 << 25);
    g1[1] = (384u & 0xFFFFu) << 16;
    g1[2] = (unsigned int)N_ << 16;
    g1[3] = (384 << 16);
    g1[4] = 16;                                  // tile_dim1=16, tile_dim2 unused (2-D)
    g1[5] = 384;
    g1[6] = 0;
    g1[7] = 0;
    asm volatile("tensor_load_to_lds %0, %1"
                 :: "s"(g0), "s"(g1)
                 : "memory");
    __builtin_amdgcn_s_wait_tensorcnt(0);
  }
  __syncthreads();

  const _Float16* WmBase = wfold + (size_t)3 * C_ * C_;
  const float*    Bm     = biasfold + 3 * C_;
  const unsigned char* lb = lds2;
  int rb = ROWB_ * lr;

  for (int round = 0; round < 3; ++round) {
    int ct = wv * 3 + round;                    // 0..23
    const _Float16* wrow = WmBase + (size_t)(ct * 16 + lr) * C_;
    v16h afr[12];
    #pragma unroll
    for (int kk = 0; kk < 12; ++kk) {
      int kb = kk * 32;
      float4* ap = reinterpret_cast<float4*>(&afr[kk]);
      ap[0] = *reinterpret_cast<const float4*>(wrow + kb + 16 * khi);
      ap[1] = *reinterpret_cast<const float4*>(wrow + kb + 16 * khi + 8);
    }
    v8f acc = {};
    #pragma unroll
    for (int kk = 0; kk < 12; ++kk) {
      int h   = kk * 32 + 16 * khi;
      int off = rb + 2 * h + 16 * (h >> 7);
      v16h bfr;
      float4* bp = reinterpret_cast<float4*>(&bfr);
      bp[0] = *reinterpret_cast<const float4*>(lb + off);
      bp[1] = *reinterpret_cast<const float4*>(lb + off + 16);
      acc = __builtin_amdgcn_wmma_f32_16x16x32_f16(false, afr[kk], false, bfr,
                                                   (short)0, acc, false, false);
    }
    #pragma unroll
    for (int r = 0; r < 8; ++r) {
      int o = ct * 16 + r + 8 * khi;
      size_t idx = ((size_t)tb * C_ + o) * N_ + nt * 16 + lr;
      out[idx] = acc[r] + Bm[o] + x[idx];
    }
  }
}

// ---------------------------------------------------------------------------
extern "C" void kernel_launch(void* const* d_in, const int* in_sizes, int n_in,
                              void* d_out, int out_size, void* d_ws, size_t ws_size,
                              hipStream_t stream)
{
  const float* x   = (const float*)d_in[0];
  const float* qw  = (const float*)d_in[1];
  const float* kw  = (const float*)d_in[2];
  const float* vw  = (const float*)d_in[3];
  const float* thw = (const float*)d_in[4];
  const float* pw  = (const float*)d_in[5];
  const float* pb  = (const float*)d_in[6];
  const float* bn[16];
  for (int i = 0; i < 16; ++i) bn[i] = (const float*)d_in[7 + i];

  char* ws = (char*)d_ws;
  _Float16* xs  = (_Float16*)(ws + XS_OFF);
  _Float16* qb  = (_Float16*)(ws + Q_OFF);
  _Float16* kb  = (_Float16*)(ws + K_OFF);
  _Float16* vb  = (_Float16*)(ws + V_OFF);
  _Float16* wf  = (_Float16*)(ws + W_OFF);
  float* biasf  = (float*)(ws + BIAS_OFF);
  float* kvraw  = (float*)(ws + KVRAW_OFF);
  float* kvs    = (float*)(ws + KVS_OFF);
  _Float16* qm  = xs;   // xs dead after qkv_gemm; reuse region for masked q

  prep_weights<<<(4 * C_ * C_ + 255) / 256, 256, 0, stream>>>(
      qw, kw, vw, pw, pb,
      bn[0], bn[1], bn[2], bn[3],
      bn[4], bn[5], bn[6], bn[7],
      bn[8], bn[9], bn[10], bn[11],
      bn[12], bn[13], bn[14], bn[15],
      wf, biasf);

  lif_transpose<<<dim3(N_ / 32, C_ / 32, B_), dim3(32, 8, 1), 0, stream>>>(x, xs);

  // 512 workgroups = 8 b x 64 n-tiles; each stages one TDM panel, 8 waves x 9 rounds
  qkv_gemm<<<512, 256, 0, stream>>>(xs, wf, biasf, qb, kb, vb);

  kv_reduce<<<32, 384, 0, stream>>>(kb, vb, kvraw);
  talk_lif<<<B_, C_, 0, stream>>>(thw, kvraw, kvs);

  size_t total = (size_t)T_ * B_ * N_ * C_;
  mask_q<<<(int)((total + 255) / 256), 256, 0, stream>>>(qb, kvs, qm);

  // 2048 workgroups = 32 tb x 64 n-tiles; 8 waves x 3 rounds over ctiles
  proj_gemm<<<2048, 256, 0, stream>>>(qm, wf, biasf, x, (float*)d_out);
}
